// GraphEmbedding_29721173688506
// MI455X (gfx1250) — matrix operand order repaired and checked by reading, not compiled, not run on hardware
//
#include <hip/hip_runtime.h>
#include <hip/hip_bf16.h>

// Problem constants (from reference)
#define NB   8      // N
#define TT   192    // T
#define HH   1024   // H
#define DD   128    // D
#define PP   4      // P
#define TP   48     // T/P
#define EE   16384  // E
#define MM   (EE + HH)   // 17408 combined edges (edges + self loops)
#define NCH  34     // MM / 512 chunks
#define CHS  512

#define LDA  136    // padded K-stride (halves) for A tile in LDS
#define LDB  136    // padded K-stride (halves) for B tile in LDS
#define LDS_STAGE 132 // padded D-stride (floats) for fp32 staging

typedef _Float16 v16h __attribute__((ext_vector_type(16)));
typedef _Float16 v8h  __attribute__((ext_vector_type(8)));
typedef float    v8f  __attribute__((ext_vector_type(8)));

union AF { v16h v; v8h p[2]; };

// ---------------------------------------------------------------------------
// CSR build (fully deterministic, integer atomics only)
// ---------------------------------------------------------------------------

// Per-chunk histogram of dst. 34 blocks x 512 threads, chunk = blockIdx.x.
__global__ void histK(const int* __restrict__ edges, int* __restrict__ hist) {
  __shared__ int lh[HH];
  const int c = blockIdx.x;
  for (int i = threadIdx.x; i < HH; i += CHS) lh[i] = 0;
  __syncthreads();
  const int e = c * CHS + threadIdx.x;           // always < MM (34*512 == MM)
  const int dst = (e < EE) ? edges[2 * e + 1] : (e - EE);
  atomicAdd(&lh[dst], 1);
  __syncthreads();
  for (int i = threadIdx.x; i < HH; i += CHS) hist[c * HH + i] = lh[i];
}

// Single block of 1024: degrees, exclusive scan -> rowptr, per-chunk starts.
__global__ void scanK(const int* __restrict__ hist, int* __restrict__ rowptr,
                      float* __restrict__ degf, int* __restrict__ chunkStart) {
  __shared__ int sc[HH];
  const int n = threadIdx.x;
  int deg = 0;
  for (int c = 0; c < NCH; ++c) deg += hist[c * HH + n];
  degf[n] = (float)deg;
  sc[n] = deg;
  __syncthreads();
  for (int off = 1; off < HH; off <<= 1) {
    int v = (n >= off) ? sc[n - off] : 0;
    __syncthreads();
    sc[n] += v;
    __syncthreads();
  }
  const int excl = sc[n] - deg;
  rowptr[n] = excl;
  if (n == HH - 1) rowptr[HH] = sc[n];
  int run = excl;
  for (int c = 0; c < NCH; ++c) { chunkStart[c * HH + n] = run; run += hist[c * HH + n]; }
}

// Inverse node-group map from node_split (4,256).
__global__ void ginvK(const int* __restrict__ node_split, int* __restrict__ ginv) {
  const int i = blockIdx.x * 256 + threadIdx.x;   // flat index into node_split
  ginv[node_split[i]] = i >> 8;                   // group = i / 256
}

// Deterministic scatter: rank within chunk = #earlier same-dst edges in chunk.
__global__ void scatterK(const int* __restrict__ edges, const int* __restrict__ chunkStart,
                         const float* __restrict__ degf,
                         int* __restrict__ col, float* __restrict__ cf) {
  __shared__ int ss[CHS], sd[CHS];
  const int c = blockIdx.x, i = threadIdx.x;
  const int e = c * CHS + i;
  int s, d;
  if (e < EE) { s = edges[2 * e]; d = edges[2 * e + 1]; } else { s = e - EE; d = s; }
  ss[i] = s; sd[i] = d;
  __syncthreads();
  int cnt = 0;
  for (int j = 0; j < i; ++j) cnt += (sd[j] == d) ? 1 : 0;
  const int pos = chunkStart[c * HH + d] + cnt;
  col[pos] = s;
  cf[pos] = rsqrtf(degf[s] * degf[d]);
}

// ---------------------------------------------------------------------------
// time_emb: (N,TP,D) conv over (hour+wday) embeddings; also writes time_rep.
// grid: 384 blocks (b*TP+t), 128 threads (o)
// ---------------------------------------------------------------------------
__global__ void timeK(const int* __restrict__ xm, const float* __restrict__ hour_emb,
                      const float* __restrict__ wday_emb, const float* __restrict__ tw,
                      const float* __restrict__ tb, float* __restrict__ time_emb,
                      float* __restrict__ out2) {
  __shared__ float es[PP][DD];
  const int r = blockIdx.x;
  const int b = r / TP, t = r - b * TP;
  const int o = threadIdx.x;
  for (int k = 0; k < PP; ++k) {
    const int tt = 4 * t + k;
    const int hi = xm[(b * TT + tt) * 2 + 1];
    const int wi = xm[(b * TT + tt) * 2 + 0];
    es[k][o] = hour_emb[hi * DD + o] + wday_emb[wi * DD + o];
  }
  __syncthreads();
  float acc = tb[o];
  for (int k = 0; k < PP; ++k) {
    const float* w = tw + (size_t)o * DD * PP + k;   // tw[o][c][k]
    #pragma unroll 4
    for (int ci = 0; ci < DD; ++ci) acc += es[k][ci] * w[ci * PP];
  }
  time_emb[(size_t)r * DD + o] = acc;
  for (int g = 0; g < 4; ++g)
    out2[((size_t)b * TT + (t * 4 + g)) * DD + o] = acc;   // time_rep (repeat x4)
}

// ---------------------------------------------------------------------------
// Kernel 1: fused [4-tap conv ∘ SpMM] -> 128x128 GEMM (W1) -> relu -> h (f16)
// grid: (16 node tiles, 384 r), 256 threads (8 waves)
// ---------------------------------------------------------------------------
__global__ __launch_bounds__(256) void gcnK1(
    const float* __restrict__ x, const int* __restrict__ rowptr,
    const int* __restrict__ col, const float* __restrict__ cf,
    const float* __restrict__ tconv_w, const float* __restrict__ tconv_b,
    const float* __restrict__ W1, const float* __restrict__ b1,
    _Float16* __restrict__ h) {
  __shared__ __align__(16) _Float16 ldsB[DD * LDB];   // W1, col-major [n][k]
  __shared__ __align__(16) _Float16 ldsA[64 * LDA];   // prop tile [node][k]
  __shared__ float sx[64][6];
  __shared__ float twl[DD * PP], tbl[DD], bl[DD];

  const int tid = threadIdx.x;
  const int tile = blockIdx.x, r = blockIdx.y;
  const int b = r / TP, t = r - b * TP;

  // Load W1 -> f16, column-major for B fragments
  for (int i = tid; i < DD * DD; i += 256) {
    const int k = i >> 7, n = i & 127;
    ldsB[n * LDB + k] = (_Float16)W1[i];
  }
  for (int i = tid; i < DD * PP; i += 256) twl[i] = tconv_w[i];
  if (tid < DD) { tbl[tid] = tconv_b[tid]; bl[tid] = b1[tid]; }

  // Phase 1: per-node edge accumulation (conv factored out: only 4+1 sums)
  if (tid < 64) {
    const int node = tile * 64 + tid;
    const float* xb = x + ((size_t)b * TT + 4 * t) * HH;
    float a0 = 0.f, a1 = 0.f, a2 = 0.f, a3 = 0.f, cs = 0.f;
    const int e0 = rowptr[node], e1 = rowptr[node + 1];
    for (int e = e0; e < e1; ++e) {
      const float cc = cf[e];
      const int s = col[e];
      a0 += cc * xb[s];
      a1 += cc * xb[HH + s];
      a2 += cc * xb[2 * HH + s];
      a3 += cc * xb[3 * HH + s];
      cs += cc;
    }
    sx[tid][0] = a0; sx[tid][1] = a1; sx[tid][2] = a2; sx[tid][3] = a3; sx[tid][4] = cs;
  }
  __syncthreads();

  // Phase 2: expand to 128-dim f16 A tile (prop = conv(sx))
  {
    const int n = tid >> 2, q = tid & 3;
    const float s0 = sx[n][0], s1 = sx[n][1], s2 = sx[n][2], s3 = sx[n][3], cs = sx[n][4];
    #pragma unroll 8
    for (int dd = 0; dd < 32; ++dd) {
      const int d = q * 32 + dd;
      const float v = twl[d * 4 + 0] * s0 + twl[d * 4 + 1] * s1 +
                      twl[d * 4 + 2] * s2 + twl[d * 4 + 3] * s3 + tbl[d] * cs;
      ldsA[n * LDA + d] = (_Float16)v;
    }
  }
  __syncthreads();

  // Phase 3: WMMA f16 -> f32. wave w: M-tile = w>>1, N-half = w&1
  const int lane = tid & 31, wv = tid >> 5;
  const int mt = wv >> 1, nh = wv & 1;
  const int lrow = lane & 15, hlf = lane >> 4;
  const v8f zero = {0.f, 0.f, 0.f, 0.f, 0.f, 0.f, 0.f, 0.f};
  v8f acc[4] = {zero, zero, zero, zero};
  #pragma unroll
  for (int k0 = 0; k0 < DD; k0 += 32) {
    AF a;
    const _Float16* ap = &ldsA[(mt * 16 + lrow) * LDA + k0 + 8 * hlf];
    a.p[0] = *(const v8h*)ap;
    a.p[1] = *(const v8h*)(ap + 16);
    #pragma unroll
    for (int j = 0; j < 4; ++j) {
      AF bf;
      const int n = nh * 64 + j * 16 + lrow;
      const _Float16* bp = &ldsB[n * LDB + k0 + 8 * hlf];
      bf.p[0] = *(const v8h*)bp;
      bf.p[1] = *(const v8h*)(bp + 16);
      acc[j] = __builtin_amdgcn_wmma_f32_16x16x32_f16(false, a.v, false, bf.v,
                                                      (short)0, acc[j], false, false);
    }
  }
  __syncthreads();

  // Epilogue: bias + relu + f16, stage via LDS for coalesced global stores
  #pragma unroll
  for (int j = 0; j < 4; ++j) {
    const int d = nh * 64 + j * 16 + lrow;
    const float bias = bl[d];
    #pragma unroll
    for (int rI = 0; rI < 8; ++rI) {
      float v = acc[j][rI] + bias;
      v = v > 0.f ? v : 0.f;
      const int nl = mt * 16 + rI + 8 * hlf;
      ldsA[nl * LDA + d] = (_Float16)v;
    }
  }
  __syncthreads();
  {
    const int row = tid >> 2, seg = tid & 3;
    _Float16* hp = h + (((size_t)r * HH) + tile * 64 + row) * DD + seg * 32;
    const _Float16* sp = &ldsA[row * LDA + seg * 32];
    #pragma unroll
    for (int u = 0; u < 4; ++u) *(v8h*)(hp + u * 8) = *(const v8h*)(sp + u * 8);
  }
}

// ---------------------------------------------------------------------------
// Kernel 2: SpMM over h (f16) -> GEMM (W2) + b2 -> per-tile group partials
// grid: (16 node tiles, 384 r), 256 threads
// ---------------------------------------------------------------------------
__global__ __launch_bounds__(256) void gcnK2(
    const _Float16* __restrict__ h, const int* __restrict__ rowptr,
    const int* __restrict__ col, const float* __restrict__ cf,
    const float* __restrict__ W2, const float* __restrict__ b2,
    const int* __restrict__ ginv, float* __restrict__ tokenPart) {
  __shared__ __align__(16) char ldsBraw[DD * LDB * 2];  // W2 f16 / fp32 stage reuse
  __shared__ __align__(16) _Float16 ldsA[64 * LDA];
  __shared__ float bl[DD];
  __shared__ int gl[64];
  _Float16* ldsB = (_Float16*)ldsBraw;

  const int tid = threadIdx.x;
  const int tile = blockIdx.x, r = blockIdx.y;

  for (int i = tid; i < DD * DD; i += 256) {
    const int k = i >> 7, n = i & 127;
    ldsB[n * LDB + k] = (_Float16)W2[i];
  }
  if (tid < DD) bl[tid] = b2[tid];
  if (tid < 64) gl[tid] = ginv[tile * 64 + tid];

  // Phase 1: SpMM gather-accumulate in fp32 (4 threads/node x 32 dims)
  {
    const int n = tid >> 2, q = tid & 3;
    float acc32[32];
    #pragma unroll
    for (int i = 0; i < 32; ++i) acc32[i] = 0.f;
    const int node = tile * 64 + n;
    const int e0 = rowptr[node], e1 = rowptr[node + 1];
    const _Float16* hr = h + (size_t)r * HH * DD + q * 32;
    for (int e = e0; e < e1; ++e) {
      const float cc = cf[e];
      const int s = col[e];
      if (e + 1 < e1) __builtin_prefetch(hr + (size_t)col[e + 1] * DD, 0, 0);
      const _Float16* hp = hr + (size_t)s * DD;
      #pragma unroll
      for (int u = 0; u < 4; ++u) {
        const v8h hv = *(const v8h*)(hp + u * 8);
        #pragma unroll
        for (int w = 0; w < 8; ++w) acc32[u * 8 + w] += cc * (float)hv[w];
      }
    }
    #pragma unroll
    for (int i = 0; i < 32; ++i) ldsA[n * LDA + q * 32 + i] = (_Float16)acc32[i];
  }
  __syncthreads();

  // Phase 2: WMMA (same tiling as K1)
  const int lane = tid & 31, wv = tid >> 5;
  const int mt = wv >> 1, nh = wv & 1;
  const int lrow = lane & 15, hlf = lane >> 4;
  const v8f zero = {0.f, 0.f, 0.f, 0.f, 0.f, 0.f, 0.f, 0.f};
  v8f acc[4] = {zero, zero, zero, zero};
  #pragma unroll
  for (int k0 = 0; k0 < DD; k0 += 32) {
    AF a;
    const _Float16* ap = &ldsA[(mt * 16 + lrow) * LDA + k0 + 8 * hlf];
    a.p[0] = *(const v8h*)ap;
    a.p[1] = *(const v8h*)(ap + 16);
    #pragma unroll
    for (int j = 0; j < 4; ++j) {
      AF bf;
      const int n = nh * 64 + j * 16 + lrow;
      const _Float16* bp = &ldsB[n * LDB + k0 + 8 * hlf];
      bf.p[0] = *(const v8h*)bp;
      bf.p[1] = *(const v8h*)(bp + 16);
      acc[j] = __builtin_amdgcn_wmma_f32_16x16x32_f16(false, a.v, false, bf.v,
                                                      (short)0, acc[j], false, false);
    }
  }
  __syncthreads();

  // Epilogue: +bias, stage fp32 (reuse ldsB region), reduce 64 nodes per group
  float* stage = (float*)ldsBraw;   // [64][LDS_STAGE]
  #pragma unroll
  for (int j = 0; j < 4; ++j) {
    const int d = nh * 64 + j * 16 + lrow;
    const float bias = bl[d];
    #pragma unroll
    for (int rI = 0; rI < 8; ++rI) {
      const int nl = mt * 16 + rI + 8 * hlf;
      stage[nl * LDS_STAGE + d] = acc[j][rI] + bias;
    }
  }
  __syncthreads();
  if (tid < DD) {
    const int d = tid;
    float gs0 = 0.f, gs1 = 0.f, gs2 = 0.f, gs3 = 0.f;
    for (int row = 0; row < 64; ++row) {
      const float v = stage[row * LDS_STAGE + d];
      const int g = gl[row];
      if (g == 0) gs0 += v; else if (g == 1) gs1 += v; else if (g == 2) gs2 += v; else gs3 += v;
    }
    float* tp = tokenPart + (((size_t)r * 4) * 16 + tile) * DD + d;
    tp[0 * 16 * DD] = gs0;
    tp[1 * 16 * DD] = gs1;
    tp[2 * 16 * DD] = gs2;
    tp[3 * 16 * DD] = gs3;
  }
}

// ---------------------------------------------------------------------------
// Final: out = mean(token partials)/256 + time_emb. grid 384 x 128
// ---------------------------------------------------------------------------
__global__ void finalK(const float* __restrict__ tokenPart, const float* __restrict__ time_emb,
                       float* __restrict__ out1) {
  const int r = blockIdx.x, d = threadIdx.x;
  const int b = r / TP, t = r - b * TP;
  const float te = time_emb[(size_t)r * DD + d];
  for (int g = 0; g < 4; ++g) {
    float s = 0.f;
    const float* tp = tokenPart + (((size_t)r * 4 + g) * 16) * DD + d;
    #pragma unroll
    for (int tl = 0; tl < 16; ++tl) s += tp[tl * DD];
    out1[((size_t)b * TT + (t * 4 + g)) * DD + d] = s * (1.0f / 256.0f) + te;
  }
}

// ---------------------------------------------------------------------------
extern "C" void kernel_launch(void* const* d_in, const int* in_sizes, int n_in,
                              void* d_out, int out_size, void* d_ws, size_t ws_size,
                              hipStream_t stream) {
  const float* x          = (const float*)d_in[0];
  const int*   x_mark     = (const int*)d_in[1];
  const int*   edges      = (const int*)d_in[2];
  const int*   node_split = (const int*)d_in[3];
  const float* hour_emb   = (const float*)d_in[4];
  const float* wday_emb   = (const float*)d_in[5];
  const float* timeconv_w = (const float*)d_in[6];
  const float* timeconv_b = (const float*)d_in[7];
  const float* tconv_w    = (const float*)d_in[8];
  const float* tconv_b    = (const float*)d_in[9];
  const float* W1         = (const float*)d_in[10];
  const float* b1         = (const float*)d_in[11];
  const float* W2         = (const float*)d_in[12];
  const float* b2         = (const float*)d_in[13];

  float* out1 = (float*)d_out;                       // (N, T, D) token+time
  float* out2 = (float*)d_out + (size_t)NB * TT * DD; // time_rep

  // Workspace carve-up (256B aligned)
  char* w = (char*)d_ws;
  size_t off = 0;
  auto take = [&](size_t bytes) { char* p = w + off; off += (bytes + 255) & ~(size_t)255; return p; };
  _Float16* hbuf      = (_Float16*)take((size_t)NB * TP * HH * DD * 2);
  int*      hist      = (int*)take((size_t)NCH * HH * 4);
  int*      chunkSt   = (int*)take((size_t)NCH * HH * 4);
  int*      rowptr    = (int*)take((size_t)(HH + 1) * 4);
  float*    degf      = (float*)take((size_t)HH * 4);
  int*      colb      = (int*)take((size_t)MM * 4);
  float*    cfb       = (float*)take((size_t)MM * 4);
  int*      ginv      = (int*)take((size_t)HH * 4);
  float*    time_emb  = (float*)take((size_t)NB * TP * DD * 4);
  float*    tokenPart = (float*)take((size_t)NB * TP * 4 * 16 * DD * 4);
  (void)ws_size; (void)n_in; (void)in_sizes; (void)out_size;

  // Deterministic CSR build
  histK<<<NCH, CHS, 0, stream>>>(edges, hist);
  scanK<<<1, HH, 0, stream>>>(hist, rowptr, degf, chunkSt);
  ginvK<<<4, 256, 0, stream>>>(node_split, ginv);
  scatterK<<<NCH, CHS, 0, stream>>>(edges, chunkSt, degf, colb, cfb);

  // time_emb + time_rep
  timeK<<<NB * TP, DD, 0, stream>>>(x_mark, hour_emb, wday_emb, timeconv_w,
                                    timeconv_b, time_emb, out2);

  // Fused conv+propagate+GEMM(W1)+relu -> h
  gcnK1<<<dim3(HH / 64, NB * TP), 256, 0, stream>>>(x, rowptr, colb, cfb,
                                                    tconv_w, tconv_b, W1, b1, hbuf);
  // propagate(h)+GEMM(W2)+bias -> grouped partials
  gcnK2<<<dim3(HH / 64, NB * TP), 256, 0, stream>>>(hbuf, rowptr, colb, cfb,
                                                    W2, b2, ginv, tokenPart);
  // mean + add time_emb
  finalK<<<NB * TP, DD, 0, stream>>>(tokenPart, time_emb, out1);
}
